// LearnedDropout_25950192402554
// MI455X (gfx1250) — matrix operand order repaired and checked by reading, not compiled, not run on hardware
//
#include <hip/hip_runtime.h>
#include <hip/hip_bf16.h>
#include <math.h>

typedef __attribute__((ext_vector_type(16))) __bf16 v16bf;
typedef __attribute__((ext_vector_type(8)))  float  v8f;

#define B_  4
#define T_  2048
#define C_  1024
#define H_  16
#define HS  64
#define BT_ (B_ * T_)     // 8192 rows
#define C3_ (3 * C_)      // 3072 cols

// ---------------------------------------------------------------------------
// WMMA fragment loaders (layouts per CDNA5 ISA 7.12.2, wave32)
// ---------------------------------------------------------------------------
__device__ __forceinline__ v16bf frag16(const __bf16* p0, const __bf16* p1) {
  union { uint4 q[2]; v16bf v; } u;
  u.q[0] = *reinterpret_cast<const uint4*>(p0);
  u.q[1] = *reinterpret_cast<const uint4*>(p1);
  return u.v;
}

// A-matrix 16x32 (row-major, stride lda elements), K-chunk base already added.
// lanes 0-15: row=lane, K={0..7,16..23}; lanes 16-31: row=lane-16, K={8..15,24..31}
__device__ __forceinline__ v16bf load_a32(const __bf16* base, int lda, int lane) {
  const __bf16* p0 = base + (lane & 15) * lda + ((lane >> 4) << 3);
  return frag16(p0, p0 + 16);
}

// B-matrix 32x16 from N-major (transposed) storage: row n contiguous in k.
// lane&15 = N; lanes 0-15 hold K={0..7,16..23}, lanes 16-31 hold K={8..15,24..31}
__device__ __forceinline__ v16bf load_bt32(const __bf16* base, int ldb, int lane) {
  const __bf16* p = base + (lane & 15) * ldb + ((lane >> 4) << 3);
  return frag16(p, p + 16);
}

__device__ __forceinline__ v8f zero8() {
  v8f z = {0.f, 0.f, 0.f, 0.f, 0.f, 0.f, 0.f, 0.f};
  return z;
}

#define WMMA_BF16(A, B, Cacc) \
  __builtin_amdgcn_wmma_f32_16x16x32_bf16(false, (A), false, (B), (short)0, (Cacc), false, false)

// ---------------------------------------------------------------------------
// K0a: x (f32) -> bf16, vectorized float4 -> 4xbf16
// ---------------------------------------------------------------------------
__global__ void cvt_x_kernel(const float4* __restrict__ x4, __bf16* __restrict__ xbf, int n4) {
  int i = blockIdx.x * blockDim.x + threadIdx.x;
  if (i < n4) {
    float4 f = x4[i];
    union { __bf16 h[4]; uint2 u; } p;
    p.h[0] = (__bf16)f.x; p.h[1] = (__bf16)f.y;
    p.h[2] = (__bf16)f.z; p.h[3] = (__bf16)f.w;
    *reinterpret_cast<uint2*>(xbf + (size_t)i * 4) = p.u;
  }
}

// K0b: w_qkv [C,3C] f32 -> wT [3C,C] bf16 (N-major so B-frags vectorize)
__global__ void cvt_wT_kernel(const float* __restrict__ w, __bf16* __restrict__ wT) {
  int i = blockIdx.x * blockDim.x + threadIdx.x;   // over (3072*1024)/4
  if (i < (C3_ * C_) / 4) {
    const int base = i * 4;
    const int nn = base >> 10;        // 0..3071
    const int kk = base & (C_ - 1);   // 0..1023, multiple of 4
    union { __bf16 h[4]; uint2 u; } p;
    #pragma unroll
    for (int j = 0; j < 4; ++j) p.h[j] = (__bf16)w[(size_t)(kk + j) * C3_ + nn];
    *reinterpret_cast<uint2*>(wT + (size_t)base) = p.u;
  }
}

// ---------------------------------------------------------------------------
// K1: qkv = x @ w_qkv  (bf16 WMMA, f32 accum). Each wave: 32x64 output tile.
// K-loop unrolled x2 with ping-pong fragment register sets: refills write the
// alternate set directly (no inter-set copies), load-to-use distance >= 8 WMMA.
// Writes q,k as [B,H,T,hs] bf16; v transposed as [B,H,hs,T] bf16.
// ---------------------------------------------------------------------------
__global__ __launch_bounds__(256) void gemm_qkv_kernel(
    const __bf16* __restrict__ xbf, const __bf16* __restrict__ wT,
    __bf16* __restrict__ qb, __bf16* __restrict__ kbuf, __bf16* __restrict__ vb)
{
  const int lane = threadIdx.x & 31;
  const int wave = threadIdx.x >> 5;
  const int m0 = (blockIdx.x * 8 + wave) * 32;   // 32 token rows per wave
  const int n0 = blockIdx.y * 64;                // col block in [0,3072)

  v8f acc[2][4];
  #pragma unroll
  for (int mi = 0; mi < 2; ++mi)
    #pragma unroll
    for (int j = 0; j < 4; ++j) acc[mi][j] = zero8();

  const __bf16* ar0 = xbf + (size_t)m0 * C_;
  const __bf16* ar1 = ar0 + (size_t)16 * C_;
  const __bf16* wTn = wT + (size_t)n0 * C_;

  // prologue: set A covers k, set B covers k+32
  v16bf a0A = load_a32(ar0, C_, lane);
  v16bf a1A = load_a32(ar1, C_, lane);
  v16bf bA[4];
  #pragma unroll
  for (int j = 0; j < 4; ++j) bA[j] = load_bt32(wTn + (size_t)(j * 16) * C_, C_, lane);
  v16bf a0B = load_a32(ar0 + 32, C_, lane);
  v16bf a1B = load_a32(ar1 + 32, C_, lane);
  v16bf bB[4];
  #pragma unroll
  for (int j = 0; j < 4; ++j) bB[j] = load_bt32(wTn + (size_t)(j * 16) * C_ + 32, C_, lane);

  #pragma unroll 1
  for (int k = 0; k < C_ - 64; k += 64) {
    // compute set A (k-chunk k), then refill set A for k+64
    #pragma unroll
    for (int j = 0; j < 4; ++j) {
      acc[0][j] = WMMA_BF16(a0A, bA[j], acc[0][j]);
      acc[1][j] = WMMA_BF16(a1A, bA[j], acc[1][j]);
    }
    __builtin_prefetch(ar0 + k + 512, 0, 1);
    __builtin_prefetch(wTn + k + 512, 0, 1);
    a0A = load_a32(ar0 + k + 64, C_, lane);
    a1A = load_a32(ar1 + k + 64, C_, lane);
    #pragma unroll
    for (int j = 0; j < 4; ++j) bA[j] = load_bt32(wTn + (size_t)(j * 16) * C_ + k + 64, C_, lane);

    // compute set B (k-chunk k+32), then refill set B for k+96
    #pragma unroll
    for (int j = 0; j < 4; ++j) {
      acc[0][j] = WMMA_BF16(a0B, bB[j], acc[0][j]);
      acc[1][j] = WMMA_BF16(a1B, bB[j], acc[1][j]);
    }
    a0B = load_a32(ar0 + k + 96, C_, lane);
    a1B = load_a32(ar1 + k + 96, C_, lane);
    #pragma unroll
    for (int j = 0; j < 4; ++j) bB[j] = load_bt32(wTn + (size_t)(j * 16) * C_ + k + 96, C_, lane);
  }
  // tail: k = C_-64 (set A) and k = C_-32 (set B)
  #pragma unroll
  for (int j = 0; j < 4; ++j) {
    acc[0][j] = WMMA_BF16(a0A, bA[j], acc[0][j]);
    acc[1][j] = WMMA_BF16(a1A, bA[j], acc[1][j]);
  }
  #pragma unroll
  for (int j = 0; j < 4; ++j) {
    acc[0][j] = WMMA_BF16(a0B, bB[j], acc[0][j]);
    acc[1][j] = WMMA_BF16(a1B, bB[j], acc[1][j]);
  }

  const int n = lane & 15;
  const int rb = (lane >> 4) * 8;
  #pragma unroll
  for (int j = 0; j < 4; ++j) {
    const int c3 = n0 + j * 16 + n;
    const int sec = c3 >> 10;          // 0=q 1=k 2=v
    const int cc = c3 & (C_ - 1);
    const int h = cc >> 6, d = cc & 63;
    #pragma unroll
    for (int mi = 0; mi < 2; ++mi) {
      #pragma unroll
      for (int r = 0; r < 8; ++r) {
        const int bt = m0 + mi * 16 + rb + r;
        const int bb = bt >> 11;         // T_ == 2048
        const int t  = bt & (T_ - 1);
        const int bh = bb * H_ + h;
        const __bf16 val = (__bf16)acc[mi][j][r];
        if (sec == 0)      qb  [(size_t)(bh * T_ + t) * HS + d] = val;
        else if (sec == 1) kbuf[(size_t)(bh * T_ + t) * HS + d] = val;
        else               vb  [(size_t)(bh * HS + d) * T_ + t] = val;
      }
    }
  }
}

// ---------------------------------------------------------------------------
// K2: causal masked attn (no softmax): logits = tril(Q K^T * hs^-0.5) @ V.
// One wave per 16-row q tile; k loop in 32-token blocks up to the diagonal.
// Pipelined: V frags for block i and K frags for block i+1 are issued around
// the LDS layout-conversion round trip so s_wait_dscnt overlaps VMEM.
// ---------------------------------------------------------------------------
__global__ __launch_bounds__(256) void attn_av_kernel(
    const __bf16* __restrict__ qb, const __bf16* __restrict__ kbuf,
    const __bf16* __restrict__ vb, float* __restrict__ logits)
{
  __shared__ __bf16 smem[8][2][16 * 32];   // double-buffered S staging per wave
  const int lane = threadIdx.x & 31;
  const int wave = threadIdx.x >> 5;
  const int bh = blockIdx.x;                 // 0..63
  const int qtile = blockIdx.y * 8 + wave;   // 0..127
  const int qbase = qtile * 16;
  const int b = bh >> 4, h = bh & 15;

  const __bf16* qrow = qb + (size_t)(bh * T_ + qbase) * HS;
  const v16bf aQ0 = load_a32(qrow,      HS, lane);
  const v16bf aQ1 = load_a32(qrow + 32, HS, lane);

  v8f o[4];
  #pragma unroll
  for (int j = 0; j < 4; ++j) o[j] = zero8();

  const int n = lane & 15;
  const int rb = (lane >> 4) * 8;
  const __bf16* kbase_p = kbuf + (size_t)bh * T_ * HS;
  const __bf16* vbase_p = vb + (size_t)bh * HS * T_;

  const int nb = (qbase + 47) >> 5;          // causal block count

  // prologue: K fragments for block 0 ([t*2+half])
  v16bf kf[4];
  #pragma unroll
  for (int t = 0; t < 2; ++t) {
    kf[t * 2 + 0] = load_bt32(kbase_p + (size_t)(t * 16) * HS,      HS, lane);
    kf[t * 2 + 1] = load_bt32(kbase_p + (size_t)(t * 16) * HS + 32, HS, lane);
  }

  #pragma unroll 1
  for (int blk = 0; blk < nb; ++blk) {
    const int kb0 = blk * 32;
    __bf16* sw = &smem[wave][blk & 1][0];

    // V fragments for this block: issue before the LDS round-trip
    v16bf bv[4];
    #pragma unroll
    for (int j = 0; j < 4; ++j)
      bv[j] = load_bt32(vbase_p + (size_t)(j * 16) * T_ + kb0, T_, lane);

    // S = Q K^T over two 16-token N tiles (K-dim = hs = 64 -> 2 wmma each)
    #pragma unroll
    for (int t = 0; t < 2; ++t) {
      v8f s = zero8();
      s = WMMA_BF16(aQ0, kf[t * 2 + 0], s);
      s = WMMA_BF16(aQ1, kf[t * 2 + 1], s);
      const int tok = kb0 + t * 16 + n;
      #pragma unroll
      for (int r = 0; r < 8; ++r) {
        const int qr = qbase + rb + r;
        const float v = (tok <= qr) ? s[r] * 0.125f : 0.0f;   // hs^-0.5, causal fill 0
        sw[(rb + r) * 32 + t * 16 + n] = (__bf16)v;
      }
    }

    // prefetch K fragments for next block during the LDS latency window
    if (blk + 1 < nb) {
      const int kn0 = kb0 + 32;
      #pragma unroll
      for (int t = 0; t < 2; ++t) {
        kf[t * 2 + 0] = load_bt32(kbase_p + (size_t)(kn0 + t * 16) * HS,      HS, lane);
        kf[t * 2 + 1] = load_bt32(kbase_p + (size_t)(kn0 + t * 16) * HS + 32, HS, lane);
      }
    }

    asm volatile("s_wait_dscnt 0" ::: "memory");   // cross-lane RAW on LDS within wave
    const v16bf aS = load_a32(sw, 32, lane);

    // O += S V  (K-dim = 32 tokens, 4 N tiles over hs)
    #pragma unroll
    for (int j = 0; j < 4; ++j) o[j] = WMMA_BF16(aS, bv[j], o[j]);
  }

  float* dst = logits + (size_t)(b * T_ + qbase) * C_ + h * HS;
  #pragma unroll
  for (int j = 0; j < 4; ++j)
    #pragma unroll
    for (int r = 0; r < 8; ++r)
      dst[(size_t)(rb + r) * C_ + j * 16 + n] = o[j][r];
}

// ---------------------------------------------------------------------------
// K3: per-row LayerNorm -> softmax -> log-scale -> dropout mask * x
// ---------------------------------------------------------------------------
__device__ __forceinline__ float block_sum(float v, float* sm) {
  const int tid = threadIdx.x;
  sm[tid] = v; __syncthreads();
  #pragma unroll
  for (int s = 128; s > 0; s >>= 1) {
    if (tid < s) sm[tid] += sm[tid + s];
    __syncthreads();
  }
  float r = sm[0]; __syncthreads();
  return r;
}
__device__ __forceinline__ float block_max(float v, float* sm) {
  const int tid = threadIdx.x;
  sm[tid] = v; __syncthreads();
  #pragma unroll
  for (int s = 128; s > 0; s >>= 1) {
    if (tid < s) sm[tid] = fmaxf(sm[tid], sm[tid + s]);
    __syncthreads();
  }
  float r = sm[0]; __syncthreads();
  return r;
}

__global__ __launch_bounds__(256) void row_post_kernel(
    const float* __restrict__ logits, const float* __restrict__ lnw,
    const float* __restrict__ x, const float* __restrict__ noise,
    float* __restrict__ out)
{
  __shared__ float sred[256];
  const int row = blockIdx.x, tid = threadIdx.x;
  const float* lrow = logits + (size_t)row * C_;

  float v[4];
  #pragma unroll
  for (int j = 0; j < 4; ++j) v[j] = lrow[tid + j * 256];

  float s = v[0] + v[1] + v[2] + v[3];
  float sq = v[0]*v[0] + v[1]*v[1] + v[2]*v[2] + v[3]*v[3];
  s  = block_sum(s,  sred);
  sq = block_sum(sq, sred);
  const float mu   = s * (1.0f / C_);
  const float var  = sq * (1.0f / C_) - mu * mu;
  const float rstd = rsqrtf(var + 1e-5f);

  float nv[4], mx = -3.402823466e38f;
  #pragma unroll
  for (int j = 0; j < 4; ++j) {
    nv[j] = (v[j] - mu) * rstd * lnw[tid + j * 256];
    mx = fmaxf(mx, nv[j]);
  }
  mx = block_max(mx, sred);

  float e[4], z = 0.0f;
  #pragma unroll
  for (int j = 0; j < 4; ++j) { e[j] = __expf(nv[j] - mx); z += e[j]; }
  z = block_sum(z, sred);
  const float invz = 1.0f / z;

  const double ee = 1.0 / 1024.0;
  const double lsd = (1.0 + sqrt(1.0 - 4.0 * ee) - 2.0 * ee) / (2.0 * ee * ee);
  const float LS = (float)lsd;
  const float RDEN = (float)(1.0 / log(lsd));

  #pragma unroll
  for (int j = 0; j < 4; ++j) {
    const float p = e[j] * invz;
    const float scaled = logf(fmaf(LS, p, 1.0f)) * RDEN;
    const float comp = 1.0f - scaled;
    const size_t idx = (size_t)row * C_ + tid + j * 256;
    // scaling + scaled == (noise >= comp) ? 1 : 0  in the forward pass
    out[idx] = x[idx] * ((noise[idx] >= comp) ? 1.0f : 0.0f);
  }
}

// ---------------------------------------------------------------------------
extern "C" void kernel_launch(void* const* d_in, const int* in_sizes, int n_in,
                              void* d_out, int out_size, void* d_ws, size_t ws_size,
                              hipStream_t stream) {
  const float* x     = (const float*)d_in[0];
  const float* w     = (const float*)d_in[1];
  const float* lnw   = (const float*)d_in[2];
  const float* noise = (const float*)d_in[3];
  float* out = (float*)d_out;

  char* ws = (char*)d_ws;
  size_t off = 0;
  __bf16* xbf = (__bf16*)(ws + off); off += (size_t)BT_ * C_ * 2;   // 16 MB
  __bf16* wT  = (__bf16*)(ws + off); off += (size_t)C3_ * C_ * 2;   //  6 MB
  __bf16* qb  = (__bf16*)(ws + off); off += (size_t)BT_ * C_ * 2;   // 16 MB
  __bf16* kb  = (__bf16*)(ws + off); off += (size_t)BT_ * C_ * 2;   // 16 MB
  __bf16* vb  = (__bf16*)(ws + off); off += (size_t)BT_ * C_ * 2;   // 16 MB
  float*  logits = (float*)(ws + off); off += (size_t)BT_ * C_ * 4; // 32 MB

  cvt_x_kernel <<<(BT_ * C_ / 4) / 256, 256, 0, stream>>>((const float4*)x, xbf, BT_ * C_ / 4);
  cvt_wT_kernel<<<(C3_ * C_ / 4) / 256, 256, 0, stream>>>(w, wT);
  gemm_qkv_kernel<<<dim3(BT_ / 256, C3_ / 64), 256, 0, stream>>>(xbf, wT, qb, kb, vb);
  attn_av_kernel<<<dim3(B_ * H_, (T_ / 16) / 8), 256, 0, stream>>>(qb, kb, vb, logits);
  row_post_kernel<<<BT_, 256, 0, stream>>>(logits, lnw, x, noise, out);
}
